// CausalSelfAttention_19018115186804
// MI455X (gfx1250) — matrix operand order repaired and checked by reading, not compiled, not run on hardware
//
#include <hip/hip_runtime.h>
#include <hip/hip_bf16.h>

// ---------------------------------------------------------------------------
// CausalSelfAttention on MI455X (gfx1250): all matmuls via v_wmma_f32_16x16x32_bf16
//   B=4, T=2048, C=1024, heads=16, head_dim=64
// ---------------------------------------------------------------------------

typedef __attribute__((ext_vector_type(16))) __bf16 v16bf;
typedef __attribute__((ext_vector_type(8)))  __bf16 v8bf;
typedef __attribute__((ext_vector_type(4)))  __bf16 v4bf;
typedef __attribute__((ext_vector_type(8)))  float  v8f;
typedef __attribute__((ext_vector_type(4)))  float  v4f;

#define DEV static __device__ __forceinline__

// A-matrix fragment (16x32 bf16, row-major source, row = lane%16):
// lo lanes: elems 0..7 -> K+0..7, elems 8..15 -> K+16..23 ; hi lanes shifted +8
DEV v16bf load_fragA(const __bf16* p, int hi8) {
  v8bf a = *(const v8bf*)(p + hi8);
  v8bf b = *(const v8bf*)(p + 16 + hi8);
  v16bf r;
#pragma unroll
  for (int i = 0; i < 8; ++i) { r[i] = a[i]; r[i + 8] = b[i]; }
  return r;
}

// B-matrix fragment (32x16 bf16, source stored [N,K]-major, col = lane%16):
// lo lanes: elems 0..7 -> K+0..7, elems 8..15 -> K+16..23 ; hi lanes shifted +8
DEV v16bf load_fragB(const __bf16* p, int hi16) {
  v8bf a = *(const v8bf*)(p + (hi16 >> 1));
  v8bf b = *(const v8bf*)(p + 16 + (hi16 >> 1));
  v16bf r;
#pragma unroll
  for (int i = 0; i < 8; ++i) { r[i] = a[i]; r[i + 8] = b[i]; }
  return r;
}

DEV v8f wmma_bf16(v16bf a, v16bf b, v8f c) {
  return __builtin_amdgcn_wmma_f32_16x16x32_bf16(false, a, false, b, (short)0, c,
                                                 false, false);
}

// ---------------------------------------------------------------------------
// Prep kernels
// ---------------------------------------------------------------------------

__global__ __launch_bounds__(256) void cvt_f32_to_bf16(
    const float* __restrict__ in, __bf16* __restrict__ out, int n4) {
  int i = blockIdx.x * blockDim.x + threadIdx.x;
  if (i >= n4) return;
  v4f x = ((const v4f*)in)[i];
  v4bf y;
#pragma unroll
  for (int k = 0; k < 4; ++k) y[k] = (__bf16)x[k];
  ((v4bf*)out)[i] = y;
}

// out[c*rows + r] = (bf16) in[r*cols + c]   (rows, cols multiples of 32)
__global__ __launch_bounds__(256) void transpose_cvt(
    const float* __restrict__ in, __bf16* __restrict__ out, int rows, int cols) {
  __shared__ float tile[32][33];
  const int cb = blockIdx.x * 32, rb = blockIdx.y * 32;
  const int tx = threadIdx.x, ty = threadIdx.y;
#pragma unroll
  for (int i = 0; i < 32; i += 8)
    tile[ty + i][tx] = in[(size_t)(rb + ty + i) * cols + cb + tx];
  __syncthreads();
#pragma unroll
  for (int i = 0; i < 32; i += 8)
    out[(size_t)(cb + ty + i) * rows + rb + tx] = (__bf16)tile[tx][ty + i];
}

// ---------------------------------------------------------------------------
// QKV GEMM: [8192,1024]bf16 @ Wt[3072,1024]bf16 -> scatter into Q/K/Vt layouts
//   Q,K : [B*16 heads, 2048, 64] bf16 ; Vt : [B*16 heads, 64, 2048] bf16
// One wave per block, 64x64 output tile, K loop step 32.
// ---------------------------------------------------------------------------
__global__ __launch_bounds__(32) void gemm_qkv(
    const __bf16* __restrict__ X, const __bf16* __restrict__ Wt,
    __bf16* __restrict__ Q, __bf16* __restrict__ K, __bf16* __restrict__ V) {
  const int lane = threadIdx.x;
  const int lr = lane & 15, hi = lane >> 4;
  const int hi8 = hi * 8, hi16 = hi * 16;
  const int m0 = blockIdx.x * 64, n0 = blockIdx.y * 64;

  const v8f zero = {0.f, 0.f, 0.f, 0.f, 0.f, 0.f, 0.f, 0.f};
  v8f acc[4][4];
#pragma unroll
  for (int i = 0; i < 4; ++i)
#pragma unroll
    for (int j = 0; j < 4; ++j) acc[i][j] = zero;

  for (int k0 = 0; k0 < 1024; k0 += 32) {
    __builtin_prefetch(X + (size_t)(m0 + lr) * 1024 + k0 + 32, 0, 1);
    __builtin_prefetch(Wt + (size_t)(n0 + lr) * 1024 + k0 + 32, 0, 1);
    v16bf af[4], bf[4];
#pragma unroll
    for (int i = 0; i < 4; ++i)
      af[i] = load_fragA(X + (size_t)(m0 + i * 16 + lr) * 1024 + k0, hi8);
#pragma unroll
    for (int j = 0; j < 4; ++j)
      bf[j] = load_fragB(Wt + (size_t)(n0 + j * 16 + lr) * 1024 + k0, hi16);
#pragma unroll
    for (int i = 0; i < 4; ++i)
#pragma unroll
      for (int j = 0; j < 4; ++j) acc[i][j] = wmma_bf16(af[i], bf[j], acc[i][j]);
  }

#pragma unroll
  for (int i = 0; i < 4; ++i) {
#pragma unroll
    for (int j = 0; j < 4; ++j) {
      const int c = n0 + j * 16 + lr;         // column in [0,3072)
      const int seg = c >> 10;                // 0=Q 1=K 2=V
      const int c1 = c & 1023;
      const int head = c1 >> 6, dd = c1 & 63;
#pragma unroll
      for (int v = 0; v < 8; ++v) {
        const int m = m0 + i * 16 + v + hi8;  // token row in [0,8192)
        const int b = m >> 11, t = m & 2047;
        const __bf16 val = (__bf16)acc[i][j][v];
        if (seg == 0)
          Q[((size_t)(b * 16 + head) * 2048 + t) * 64 + dd] = val;
        else if (seg == 1)
          K[((size_t)(b * 16 + head) * 2048 + t) * 64 + dd] = val;
        else
          V[((size_t)(b * 16 + head) * 64 + dd) * 2048 + t] = val;
      }
    }
  }
}

// ---------------------------------------------------------------------------
// Flash attention, one wave per 16-query tile. Computes S^T = K*Q^T so the
// softmax'd probabilities land exactly in the A-fragment layout for P*V.
// ---------------------------------------------------------------------------
__global__ __launch_bounds__(32) void attn(
    const __bf16* __restrict__ Q, const __bf16* __restrict__ Kx,
    const __bf16* __restrict__ Vt, __bf16* __restrict__ O) {
  const int lane = threadIdx.x;
  const int lr = lane & 15, hi = lane >> 4;
  const int hi8 = hi * 8, hi16 = hi * 16;

  const int idx = blockIdx.x;
  const int qt = idx & 127;
  const int head = (idx >> 7) & 15;
  const int b = idx >> 11;
  const int qt16 = qt * 16;

  const __bf16* Qh = Q + (size_t)(b * 16 + head) * 2048 * 64;
  const __bf16* Kh = Kx + (size_t)(b * 16 + head) * 2048 * 64;
  const __bf16* Vh = Vt + (size_t)(b * 16 + head) * 64 * 2048;

  // B-operand fragments of Q^T (d x 16 queries), d-chunks {0..31},{32..63}
  const v16bf qf0 = load_fragB(Qh + (size_t)(qt16 + lr) * 64 + 0, hi16);
  const v16bf qf1 = load_fragB(Qh + (size_t)(qt16 + lr) * 64 + 32, hi16);

  const v8f zero = {0.f, 0.f, 0.f, 0.f, 0.f, 0.f, 0.f, 0.f};
  v8f o[4];
#pragma unroll
  for (int j = 0; j < 4; ++j) o[j] = zero;

  float m_run = -__builtin_inff();
  float s_run = 0.0f;
  const float sc = 0.125f;  // 1/sqrt(64)
  const int qg = qt16 + lr; // this lane's query index
  const int kend = qt16 + 16;

  for (int kb = 0; kb < kend; kb += 32) {
    // S^T tiles: st0 -> keys kb..kb+15, st1 -> keys kb+16..kb+31 (rows),
    // columns = 16 queries (lane = query)
    v8f st0 = zero, st1 = zero;
    {
      const __bf16* kr = Kh + (size_t)(kb + lr) * 64;
      st0 = wmma_bf16(load_fragA(kr, hi8), qf0, st0);
      st0 = wmma_bf16(load_fragA(kr + 32, hi8), qf1, st0);
    }
    {
      const __bf16* kr = Kh + (size_t)(kb + 16 + lr) * 64;
      st1 = wmma_bf16(load_fragA(kr, hi8), qf0, st1);
      st1 = wmma_bf16(load_fragA(kr + 32, hi8), qf1, st1);
    }

    // scale + causal mask + local max
    float p0[8], p1[8];
    float lm = -__builtin_inff();
#pragma unroll
    for (int v = 0; v < 8; ++v) {
      const int k0i = kb + v + hi8;
      const int k1i = kb + 16 + v + hi8;
      float s0 = st0[v] * sc;
      float s1 = st1[v] * sc;
      if (k0i > qg) s0 = -__builtin_inff();
      if (k1i > qg) s1 = -__builtin_inff();
      p0[v] = s0; p1[v] = s1;
      lm = fmaxf(lm, fmaxf(s0, s1));
    }
    const float tm = fmaxf(lm, __shfl_xor(lm, 16, 32));
    const float mnew = fmaxf(m_run, tm);
    const float corr = __expf(m_run - mnew);

    float ps = 0.0f;
#pragma unroll
    for (int v = 0; v < 8; ++v) {
      p0[v] = __expf(p0[v] - mnew);
      p1[v] = __expf(p1[v] - mnew);
      ps += p0[v] + p1[v];
    }
    s_run = s_run * corr + ps + __shfl_xor(ps, 16, 32);
    m_run = mnew;

    // rescale O accumulators: row r scale lives in lane r / r+16
#pragma unroll
    for (int v = 0; v < 8; ++v) {
      const float cv = __shfl(corr, v + hi8, 32);
      o[0][v] *= cv; o[1][v] *= cv; o[2][v] *= cv; o[3][v] *= cv;
    }

    // P as A-fragment (16 queries x 32 keys) — layout matches S^T directly
    v16bf pf;
#pragma unroll
    for (int v = 0; v < 8; ++v) {
      pf[v] = (__bf16)p0[v];
      pf[v + 8] = (__bf16)p1[v];
    }

    // O += P * V  (V^T stored [d, T] so B fragments are contiguous along keys)
#pragma unroll
    for (int j = 0; j < 4; ++j) {
      const v16bf vf = load_fragB(Vh + (size_t)(j * 16 + lr) * 2048 + kb, hi16);
      o[j] = wmma_bf16(pf, vf, o[j]);
    }
  }

  // normalize rows by running sum and store token-major bf16 [8192, 1024]
  const float inv = 1.0f / s_run;
#pragma unroll
  for (int v = 0; v < 8; ++v) {
    const float iv = __shfl(inv, v + hi8, 32);
    const int t = qt16 + v + hi8;
    const size_t rowbase = ((size_t)(b * 2048 + t)) * 1024 + head * 64;
#pragma unroll
    for (int j = 0; j < 4; ++j)
      O[rowbase + j * 16 + lr] = (__bf16)(o[j][v] * iv);
  }
}

// ---------------------------------------------------------------------------
// Output projection: O_bf[8192,1024] @ Wt_proj[1024,1024] -> fp32 d_out
// ---------------------------------------------------------------------------
__global__ __launch_bounds__(32) void gemm_proj(
    const __bf16* __restrict__ A, const __bf16* __restrict__ Wt,
    float* __restrict__ out) {
  const int lane = threadIdx.x;
  const int lr = lane & 15, hi = lane >> 4;
  const int hi8 = hi * 8, hi16 = hi * 16;
  const int m0 = blockIdx.x * 64, n0 = blockIdx.y * 64;

  const v8f zero = {0.f, 0.f, 0.f, 0.f, 0.f, 0.f, 0.f, 0.f};
  v8f acc[4][4];
#pragma unroll
  for (int i = 0; i < 4; ++i)
#pragma unroll
    for (int j = 0; j < 4; ++j) acc[i][j] = zero;

  for (int k0 = 0; k0 < 1024; k0 += 32) {
    __builtin_prefetch(A + (size_t)(m0 + lr) * 1024 + k0 + 32, 0, 1);
    v16bf af[4], bf[4];
#pragma unroll
    for (int i = 0; i < 4; ++i)
      af[i] = load_fragA(A + (size_t)(m0 + i * 16 + lr) * 1024 + k0, hi8);
#pragma unroll
    for (int j = 0; j < 4; ++j)
      bf[j] = load_fragB(Wt + (size_t)(n0 + j * 16 + lr) * 1024 + k0, hi16);
#pragma unroll
    for (int i = 0; i < 4; ++i)
#pragma unroll
      for (int j = 0; j < 4; ++j) acc[i][j] = wmma_bf16(af[i], bf[j], acc[i][j]);
  }

#pragma unroll
  for (int i = 0; i < 4; ++i)
#pragma unroll
    for (int j = 0; j < 4; ++j) {
      const int c = n0 + j * 16 + lr;
#pragma unroll
      for (int v = 0; v < 8; ++v) {
        const int m = m0 + i * 16 + v + hi8;
        out[(size_t)m * 1024 + c] = acc[i][j][v];
      }
    }
}

// ---------------------------------------------------------------------------
// Launch
// ---------------------------------------------------------------------------
extern "C" void kernel_launch(void* const* d_in, const int* in_sizes, int n_in,
                              void* d_out, int out_size, void* d_ws,
                              size_t ws_size, hipStream_t stream) {
  const float* x = (const float*)d_in[0];      // [4,2048,1024]
  const float* w_qkv = (const float*)d_in[1];  // [1024,3072]
  const float* w_proj = (const float*)d_in[2]; // [1024,1024]
  float* out = (float*)d_out;                  // [4,2048,1024]

  char* ws = (char*)d_ws;
  __bf16* Xb = (__bf16*)(ws);                          // 16 MiB
  __bf16* Wq = (__bf16*)(ws + (16ull << 20));          //  6 MiB  Wt_qkv[3072,1024]
  __bf16* Wp = (__bf16*)(ws + (22ull << 20));          //  2 MiB  Wt_proj[1024,1024]
  __bf16* Qb = (__bf16*)(ws + (24ull << 20));          // 16 MiB
  __bf16* Kb = (__bf16*)(ws + (40ull << 20));          // 16 MiB
  __bf16* Vb = (__bf16*)(ws + (56ull << 20));          // 16 MiB (transposed per head)
  __bf16* Ob = (__bf16*)(ws + (72ull << 20));          // 16 MiB -> 88 MiB total

  // 1) convert activations to bf16
  cvt_f32_to_bf16<<<8192, 256, 0, stream>>>(x, Xb, (8192 * 1024) / 4);
  // 2) transpose + convert weights to [N,K] bf16
  transpose_cvt<<<dim3(96, 32), dim3(32, 8), 0, stream>>>(w_qkv, Wq, 1024, 3072);
  transpose_cvt<<<dim3(32, 32), dim3(32, 8), 0, stream>>>(w_proj, Wp, 1024, 1024);
  // 3) QKV projection -> per-head Q, K, V^T
  gemm_qkv<<<dim3(128, 48), 32, 0, stream>>>(Xb, Wq, Qb, Kb, Vb);
  // 4) flash attention, one wave per (batch, head, 16-query tile)
  attn<<<4 * 16 * 128, 32, 0, stream>>>(Qb, Kb, Vb, Ob);
  // 5) output projection -> fp32
  gemm_proj<<<dim3(128, 16), 32, 0, stream>>>(Ob, Wp, out);
}
